// TransformerEncoder_22204980920992
// MI455X (gfx1250) — compile-verified, matrix operand order checked by reading, not run on hardware
//
#include <hip/hip_runtime.h>
#include <math.h>

// Transformer encoder forward for MI455X (gfx1250): bf16 WMMA GEMMs (f32 accum)
// with Tensor-Data-Mover (TDM) async tile staging + double-buffered LDS.
// L=12, B=4, N=512, D=768, H=12, DH=64, DFF=3072.

static constexpr int L_   = 12;
static constexpr int B_   = 4;
static constexpr int N_   = 512;
static constexpr int D_   = 768;
static constexpr int H_   = 12;
static constexpr int DH_  = 64;
static constexpr int DFF_ = 3072;
static constexpr int MN_  = B_ * N_;        // 2048 token rows
static constexpr float SCALE_ = 0.125f;     // 1/sqrt(64)

typedef __attribute__((ext_vector_type(16))) __bf16 v16bf;
typedef __attribute__((ext_vector_type(8)))  float  v8f;
typedef __attribute__((ext_vector_type(4)))  unsigned int u32x4;
typedef __attribute__((ext_vector_type(8)))  int i32x8;
typedef __attribute__((ext_vector_type(4)))  int i32x4;

enum { EPI_QKV = 0, EPI_SCORES = 1, EPI_AV = 2, EPI_BIAS_RES = 3, EPI_BIAS_GELU = 4 };

// Native bf16 convert (RTNE) -> v_cvt_*bf16_f32 on gfx1250.
__device__ __forceinline__ unsigned short f2bf(float f) {
  union { __bf16 h; unsigned short u; } cv;
  cv.h = (__bf16)f;
  return cv.u;
}
__device__ __forceinline__ unsigned pack2bf(float lo, float hi) {
  return (unsigned)f2bf(lo) | ((unsigned)f2bf(hi) << 16);
}

union FragU { v16bf v; uint4 q[2]; };

// Build a 16x32 bf16 A/B fragment from an LDS tile stored [row][k] (stride in
// halfwords). Per ISA 7.12.2: lane<16 -> row=lane, K 0-7 (v0..3) + 16-23 (v4..7);
// lane>=16 -> row=lane-16, K 8-15 + 24-31. Two ds_load_b128 per fragment.
__device__ __forceinline__ v16bf frag_from_lds(const unsigned short* base, int row0, int stride) {
  int lane = threadIdx.x & 31;
  int r  = row0 + (lane & 15);
  int kb = (lane & 16) ? 8 : 0;
  const unsigned short* p = base + r * stride + kb;
  FragU u;
  u.q[0] = *(const uint4*)(p);        // K kb..kb+7
  u.q[1] = *(const uint4*)(p + 16);   // K kb+16..kb+23
  return u.v;
}

__device__ __forceinline__ v8f wmma_bf16(v16bf a, v16bf b, v8f c) {
#if defined(__HIP_DEVICE_COMPILE__)
  return __builtin_amdgcn_wmma_f32_16x16x32_bf16(false, a, false, b, (short)0, c, false, false);
#else
  return c;
#endif
}

// TDM: async DMA of a 2D bf16 tile [tile_rows x tile_k] (row stride
// row_stride_elems) from global to LDS byte offset lds_off. D# layout per
// CDNA5 ISA 8.3/8.4: group0 {count=1, lds_addr, global_addr, type=2},
// group1 {data_size=1(2B), tensor_dim0/1=1<<20 (tiles are interior),
// tile_dim0=tile_k, tile_dim1=tile_rows, tensor_dim0_stride}.
// Tracked by TENSORcnt (s_wait_tensorcnt).
__device__ __forceinline__ void tdm_load_tile_bf16(
    unsigned lds_off, const void* gptr, int tile_k, int tile_rows,
    int row_stride_elems) {
#if defined(__HIP_DEVICE_COMPILE__)
  unsigned long long ga = (unsigned long long)(uintptr_t)gptr;
  u32x4 g0;
  g0[0] = 1u;                                           // count=1 (valid user D#)
  g0[1] = lds_off;                                      // LDS byte address
  g0[2] = (unsigned)ga;                                 // global_addr[31:0]
  g0[3] = ((unsigned)(ga >> 32) & 0x01FFFFFFu) | (2u << 30);  // addr[56:32], type=2
  i32x8 g1;
  g1[0] = 0x00010000;                    // workgroup_mask=0, data_size=1 (2 bytes)
  g1[1] = 0;                             // abar=0, tensor_dim0[15:0] (1<<20 -> 0)
  g1[2] = 0x10;                          // tensor_dim0[31:16]=0x10, tensor_dim1 lo=0
  g1[3] = 0x10 | (tile_k << 16);         // tensor_dim1 hi=0x10, tile_dim0
  g1[4] = tile_rows & 0xFFFF;            // tile_dim1, tile_dim2=0
  g1[5] = row_stride_elems;              // tensor_dim0_stride[31:0]
  g1[6] = 0;
  g1[7] = 0;
  i32x4 z4 = {0, 0, 0, 0};
#if __clang_major__ >= 23
  i32x8 z8 = {0, 0, 0, 0, 0, 0, 0, 0};
  __builtin_amdgcn_tensor_load_to_lds(g0, g1, z4, z4, z8, 0);
#else
  __builtin_amdgcn_tensor_load_to_lds(g0, g1, z4, z4, 0);
#endif
#endif
}

__device__ __forceinline__ void tdm_wait0() {
#if defined(__HIP_DEVICE_COMPILE__)
  __builtin_amdgcn_s_wait_tensorcnt(0);
#endif
}

// fp32 [K,N] weight tile -> 16 regs (coalesced along n, independent loads).
__device__ __forceinline__ void wload(const float* Bp, int ldb, int k0, int n0,
                                      int t, int Kd, float r[16]) {
#pragma unroll
  for (int pass = 0; pass < 2; pass++) {
    int c  = pass * 256 + t;                // 512 chunks of 8 k-values
    int nn = c & 127;
    int kc = (c >> 7) * 8;                  // 0,8,16,24
    const float* gp = &Bp[(long)(k0 + kc) * ldb + n0 + nn];
#pragma unroll
    for (int j = 0; j < 8; j++) r[pass * 8 + j] = gp[(long)j * ldb];
    if (k0 + 32 < Kd) __builtin_prefetch(gp + (long)32 * ldb, 0, 1);
  }
}

// pack 16 regs -> bf16 and store transposed [n][k] into LDS (two b128 stores).
__device__ __forceinline__ void wstore(unsigned short* BsD, int t, const float r[16]) {
#pragma unroll
  for (int pass = 0; pass < 2; pass++) {
    int c  = pass * 256 + t;
    int nn = c & 127;
    int kc = (c >> 7) * 8;
    const float* q = r + pass * 8;
    uint4 pk;
    pk.x = pack2bf(q[0], q[1]);
    pk.y = pack2bf(q[2], q[3]);
    pk.z = pack2bf(q[4], q[5]);
    pk.w = pack2bf(q[6], q[7]);
    *(uint4*)&BsD[nn * 32 + kc] = pk;
  }
}

// Generic WMMA GEMM: C[M,N] = A[M,K](bf16, row-major) * B (+ fused epilogue).
// BDIRECT=true : Bsrc is bf16 stored [N,K] row-major -> TDM-staged directly.
// BDIRECT=false: Bsrc is fp32 [K,N] (weights); register transpose+convert.
// A tiles always TDM-staged. Double-buffered LDS (32KB dynamic, base offset 0).
// Block tile 128x128, BK=32, 256 threads = 8 waves (4x2), 32x64 per wave
// (2x4 WMMA tiles: 6 fragment loads per 8 WMMAs).
template <int EPI, typename TB, bool BDIRECT>
__global__ __launch_bounds__(256) void gemm_wmma(
    const unsigned short* __restrict__ A, int lda, long batchStrideA,
    const TB* __restrict__ Bsrc, int ldb, long batchStrideB,
    int Md, int Nd, int Kd,
    const float* __restrict__ bias,
    const float* __restrict__ resid,
    float* __restrict__ outF, long strideOutF,
    unsigned short* __restrict__ outH,
    unsigned short* __restrict__ qOut, unsigned short* __restrict__ kOut,
    unsigned short* __restrict__ vOut, float scale) {
  constexpr int BM = 128, BN = 128, BK = 32;
  constexpr int ASZ = BM * BK;            // elements per A buffer (8KB)
  constexpr int BSZ = BN * BK;            // elements per B buffer (8KB)
  extern __shared__ unsigned short smem[];            // dynamic LDS: base offset 0
  // layout (elements): [A0: ASZ][A1: ASZ][B0: BSZ][B1: BSZ]

  const int bz = blockIdx.z;
  const unsigned short* Ab = A + (long)bz * batchStrideA;
  const TB* Bp = Bsrc + (long)bz * batchStrideB;
  const int m0 = blockIdx.y * BM;
  const int n0 = blockIdx.x * BN;
  const int t  = threadIdx.x;
  const int wave = t >> 5, lane = t & 31;
  const int wr = wave >> 1, wc = wave & 1;   // 4 x 2 wave grid, 32x64 per wave
  const int brows = (Nd - n0 > BN) ? BN : (Nd - n0);   // BDIRECT row clamp (AV: 64)

  v8f acc[2][4];
  v8f zero = {0.f, 0.f, 0.f, 0.f, 0.f, 0.f, 0.f, 0.f};
#pragma unroll
  for (int i = 0; i < 2; i++)
#pragma unroll
    for (int j = 0; j < 4; j++) acc[i][j] = zero;

  const int KT = Kd / BK;
  float rB[16];

  // ---- prologue: stage tile 0 into buffer 0 ----
  if (wave == 0) {
    tdm_load_tile_bf16(0u, &Ab[(long)m0 * lda], BK, BM, lda);
    if constexpr (BDIRECT)
      tdm_load_tile_bf16((unsigned)(2 * ASZ * 2), &Bp[(long)n0 * ldb], BK, brows, ldb);
  }
  if constexpr (!BDIRECT) {
    wload((const float*)Bp, ldb, 0, n0, t, Kd, rB);
    wstore(smem + 2 * ASZ, t, rB);
  }
  if (wave == 0) tdm_wait0();
  __syncthreads();

  for (int kt = 0; kt < KT; kt++) {
    const int cur = kt & 1;
    const int nxt = cur ^ 1;
    const bool have_next = (kt + 1 < KT);
    const int kn = (kt + 1) * BK;
    // issue async staging of next tile into the other buffer
    if (have_next && wave == 0) {
      tdm_load_tile_bf16((unsigned)(nxt * ASZ * 2), &Ab[(long)m0 * lda + kn], BK, BM, lda);
      if constexpr (BDIRECT)
        tdm_load_tile_bf16((unsigned)((2 * ASZ + nxt * BSZ) * 2),
                           &Bp[(long)n0 * ldb + kn], BK, brows, ldb);
    }
    if constexpr (!BDIRECT)
      if (have_next) wload((const float*)Bp, ldb, kn, n0, t, Kd, rB);

    // compute on current buffer (overlaps TDM + weight loads)
    unsigned short* Acur = smem + cur * ASZ;
    unsigned short* Bcur = smem + 2 * ASZ + cur * BSZ;
    v16bf a0 = frag_from_lds(Acur, wr * 32 + 0,  BK);
    v16bf a1 = frag_from_lds(Acur, wr * 32 + 16, BK);
    v16bf b0 = frag_from_lds(Bcur, wc * 64 + 0,  BK);
    v16bf b1 = frag_from_lds(Bcur, wc * 64 + 16, BK);
    v16bf b2 = frag_from_lds(Bcur, wc * 64 + 32, BK);
    v16bf b3 = frag_from_lds(Bcur, wc * 64 + 48, BK);
    acc[0][0] = wmma_bf16(a0, b0, acc[0][0]);
    acc[0][1] = wmma_bf16(a0, b1, acc[0][1]);
    acc[0][2] = wmma_bf16(a0, b2, acc[0][2]);
    acc[0][3] = wmma_bf16(a0, b3, acc[0][3]);
    acc[1][0] = wmma_bf16(a1, b0, acc[1][0]);
    acc[1][1] = wmma_bf16(a1, b1, acc[1][1]);
    acc[1][2] = wmma_bf16(a1, b2, acc[1][2]);
    acc[1][3] = wmma_bf16(a1, b3, acc[1][3]);

    if constexpr (!BDIRECT)
      if (have_next) wstore(smem + 2 * ASZ + nxt * BSZ, t, rB);
    if (wave == 0 && have_next) tdm_wait0();
    __syncthreads();
  }

  // Epilogue: C/D layout (ISA 7.12.2): VGPR i, lane -> M = i + (lane&16 ? 8:0),
  // N = lane&15 within each 16x16 tile.
  const int m_base = m0 + wr * 32;
  const int n_base = n0 + wc * 64;
#pragma unroll
  for (int ti = 0; ti < 2; ti++) {
#pragma unroll
    for (int tj = 0; tj < 4; tj++) {
      v8f c = acc[ti][tj];
      int mrow = m_base + ti * 16 + ((lane & 16) ? 8 : 0);
      int ncol = n_base + tj * 16 + (lane & 15);
#pragma unroll
      for (int i = 0; i < 8; i++) {
        int m = mrow + i;
        int n = ncol;
        float v = c[i];
        if constexpr (EPI == EPI_QKV) {
          int b = m >> 9, tok = m & (N_ - 1);
          int which = n / D_;
          int r = n - which * D_;
          int h = r >> 6, dh = r & 63;
          if (which == 0) {        // Q: [b,h,tok,dh]
            qOut[((long)((b * H_ + h) * N_) + tok) * DH_ + dh] = f2bf(v);
          } else if (which == 1) { // K: [b,h,key,dh] (BDIRECT for scores)
            kOut[((long)((b * H_ + h) * N_) + tok) * DH_ + dh] = f2bf(v);
          } else {                 // V^T: [b,h,dh,key] (BDIRECT for attn*V)
            vOut[((long)((b * H_ + h) * DH_) + dh) * N_ + tok] = f2bf(v);
          }
        } else if constexpr (EPI == EPI_SCORES) {
          outF[(long)bz * strideOutF + (long)m * N_ + n] = v * scale;
        } else if constexpr (EPI == EPI_AV) {
          if (n < Nd) {
            int b = bz / H_, h = bz - b * H_;
            outH[((long)(b * N_ + m)) * D_ + h * DH_ + n] = f2bf(v);
          }
        } else if constexpr (EPI == EPI_BIAS_RES) {
          outF[(long)m * Nd + n] = v + bias[n] + resid[(long)m * Nd + n];
        } else { // EPI_BIAS_GELU (exact erf GELU)
          float g = v + bias[n];
          g = 0.5f * g * (1.0f + erff(g * 0.70710678118654752f));
          outH[(long)m * Nd + n] = f2bf(g);
        }
      }
    }
  }
  (void)Md;
}

// LayerNorm over D=768, fp32 in -> bf16 out. One block per row.
__global__ __launch_bounds__(256) void ln_bf16_kernel(
    const float* __restrict__ x, const float* __restrict__ s,
    const float* __restrict__ b, unsigned short* __restrict__ out) {
  __shared__ float red[256];
  int t = threadIdx.x;
  long row = blockIdx.x;
  const float* xr = x + row * D_;
  float v0 = xr[t], v1 = xr[t + 256], v2 = xr[t + 512];
  red[t] = v0 + v1 + v2;
  __syncthreads();
  for (int off = 128; off > 0; off >>= 1) {
    if (t < off) red[t] += red[t + off];
    __syncthreads();
  }
  float mean = red[0] * (1.0f / 768.0f);
  __syncthreads();
  float d0 = v0 - mean, d1 = v1 - mean, d2 = v2 - mean;
  red[t] = d0 * d0 + d1 * d1 + d2 * d2;
  __syncthreads();
  for (int off = 128; off > 0; off >>= 1) {
    if (t < off) red[t] += red[t + off];
    __syncthreads();
  }
  float rstd = rsqrtf(red[0] * (1.0f / 768.0f) + 1e-5f);
  unsigned short* o = out + row * D_;
  o[t]       = f2bf(d0 * rstd * s[t]       + b[t]);
  o[t + 256] = f2bf(d1 * rstd * s[t + 256] + b[t + 256]);
  o[t + 512] = f2bf(d2 * rstd * s[t + 512] + b[t + 512]);
}

// Softmax over N=512, in-place on fp32 scores (in d_out) + bf16 copy for WMMA.
__global__ __launch_bounds__(256) void softmax_kernel(
    float* __restrict__ sc, unsigned short* __restrict__ attn_bf) {
  __shared__ float red[256];
  int t = threadIdx.x;
  long row = blockIdx.x;
  float* p = sc + row * N_;
  float a = p[t], c = p[t + 256];
  red[t] = fmaxf(a, c);
  __syncthreads();
  for (int off = 128; off > 0; off >>= 1) {
    if (t < off) red[t] = fmaxf(red[t], red[t + off]);
    __syncthreads();
  }
  float mx = red[0];
  __syncthreads();
  float e0 = __expf(a - mx), e1 = __expf(c - mx);
  red[t] = e0 + e1;
  __syncthreads();
  for (int off = 128; off > 0; off >>= 1) {
    if (t < off) red[t] += red[t + off];
    __syncthreads();
  }
  float inv = 1.0f / red[0];
  float r0 = e0 * inv, r1 = e1 * inv;
  p[t] = r0;
  p[t + 256] = r1;
  unsigned short* ab = attn_bf + row * N_;
  ab[t] = f2bf(r0);
  ab[t + 256] = f2bf(r1);
}

extern "C" void kernel_launch(void* const* d_in, const int* in_sizes, int n_in,
                              void* d_out, int out_size, void* d_ws, size_t ws_size,
                              hipStream_t stream) {
  const float* x_in  = (const float*)d_in[0];
  const float* ln1_s = (const float*)d_in[1];
  const float* ln1_b = (const float*)d_in[2];
  const float* wqkv  = (const float*)d_in[3];
  const float* wo    = (const float*)d_in[4];
  const float* bo    = (const float*)d_in[5];
  const float* ln2_s = (const float*)d_in[6];
  const float* ln2_b = (const float*)d_in[7];
  const float* w1    = (const float*)d_in[8];
  const float* b1    = (const float*)d_in[9];
  const float* w2    = (const float*)d_in[10];
  const float* b2    = (const float*)d_in[11];

  // Workspace carve-up (~66 MB), 256B-aligned sub-buffers.
  char* p = (char*)d_ws;
  auto take = [&](size_t bytes) -> char* {
    char* r = p;
    p += (bytes + 255) & ~(size_t)255;
    return r;
  };
  float* x0 = (float*)take(sizeof(float) * MN_ * D_);
  float* x1 = (float*)take(sizeof(float) * MN_ * D_);
  unsigned short* h_bf    = (unsigned short*)take(2ull * MN_ * D_);
  unsigned short* q_bf    = (unsigned short*)take(2ull * B_ * H_ * N_ * DH_);
  unsigned short* k_bf    = (unsigned short*)take(2ull * B_ * H_ * N_ * DH_);
  unsigned short* v_bf    = (unsigned short*)take(2ull * B_ * H_ * N_ * DH_); // V^T [b,h,dh,key]
  unsigned short* attn_bf = (unsigned short*)take(2ull * B_ * H_ * N_ * N_);
  unsigned short* o_bf    = (unsigned short*)take(2ull * MN_ * D_);
  unsigned short* ff_bf   = (unsigned short*)take(2ull * MN_ * DFF_);

  float* out_x = (float*)d_out;
  float* out_w = out_x + (size_t)B_ * N_ * D_;   // [L,B,H,N,N] region

  constexpr size_t GEMM_LDS = (2 * 128 * 32 + 2 * 128 * 32) * sizeof(unsigned short); // 32768B

  (void)hipMemcpyAsync(x0, x_in, sizeof(float) * MN_ * D_, hipMemcpyDeviceToDevice, stream);

  for (int l = 0; l < L_; l++) {
    // LN1 -> bf16
    ln_bf16_kernel<<<MN_, 256, 0, stream>>>(x0, ln1_s + l * D_, ln1_b + l * D_, h_bf);
    // QKV: [2048,768] x [768,2304], scatter to per-head Q/K/V^T (bf16)
    gemm_wmma<EPI_QKV, float, false><<<dim3((3 * D_) / 128, MN_ / 128, 1), 256, GEMM_LDS, stream>>>(
        h_bf, D_, 0, wqkv + (long)l * D_ * 3 * D_, 3 * D_, 0,
        MN_, 3 * D_, D_, nullptr, nullptr, nullptr, 0, nullptr, q_bf, k_bf, v_bf, 0.f);
    // Scores = Q K^T * scale -> directly into d_out weights[l]
    float* sc = out_w + (long)l * B_ * H_ * N_ * N_;
    gemm_wmma<EPI_SCORES, unsigned short, true><<<dim3(N_ / 128, N_ / 128, B_ * H_), 256, GEMM_LDS, stream>>>(
        q_bf, DH_, (long)N_ * DH_, k_bf, DH_, (long)N_ * DH_,
        N_, N_, DH_, nullptr, nullptr, sc, (long)N_ * N_, nullptr, nullptr, nullptr, nullptr, SCALE_);
    // Softmax in-place on d_out + bf16 copy
    softmax_kernel<<<B_ * H_ * N_, 256, 0, stream>>>(sc, attn_bf);
    // O = attn x V (B = V^T, TDM direct staging), gather heads -> [token,768] bf16
    gemm_wmma<EPI_AV, unsigned short, true><<<dim3(1, N_ / 128, B_ * H_), 256, GEMM_LDS, stream>>>(
        attn_bf, N_, (long)N_ * N_, v_bf, N_, (long)N_ * DH_,
        N_, DH_, N_, nullptr, nullptr, nullptr, 0, o_bf, nullptr, nullptr, nullptr, 0.f);
    // x1 = x0 + O wo + bo
    gemm_wmma<EPI_BIAS_RES, float, false><<<dim3(D_ / 128, MN_ / 128, 1), 256, GEMM_LDS, stream>>>(
        o_bf, D_, 0, wo + (long)l * D_ * D_, D_, 0,
        MN_, D_, D_, bo + l * D_, x0, x1, 0, nullptr, nullptr, nullptr, nullptr, 0.f);
    // LN2 -> bf16
    ln_bf16_kernel<<<MN_, 256, 0, stream>>>(x1, ln2_s + l * D_, ln2_b + l * D_, h_bf);
    // FF1 + exact GELU -> bf16
    gemm_wmma<EPI_BIAS_GELU, float, false><<<dim3(DFF_ / 128, MN_ / 128, 1), 256, GEMM_LDS, stream>>>(
        h_bf, D_, 0, w1 + (long)l * D_ * DFF_, DFF_, 0,
        MN_, DFF_, D_, b1 + l * DFF_, nullptr, nullptr, 0, ff_bf, nullptr, nullptr, nullptr, 0.f);
    // x0 = x1 + FF w2 + b2
    gemm_wmma<EPI_BIAS_RES, float, false><<<dim3(D_ / 128, MN_ / 128, 1), 256, GEMM_LDS, stream>>>(
        ff_bf, DFF_, 0, w2 + (long)l * DFF_ * D_, D_, 0,
        MN_, D_, DFF_, b2 + l * D_, x1, x0, 0, nullptr, nullptr, nullptr, nullptr, 0.f);
  }
  (void)hipMemcpyAsync(d_out, x0, sizeof(float) * MN_ * D_, hipMemcpyDeviceToDevice, stream);
  (void)in_sizes; (void)n_in; (void)out_size; (void)ws_size;
}
